// SelfAttention_46660524704153
// MI455X (gfx1250) — matrix operand order repaired and checked by reading, not compiled, not run on hardware
//
#include <hip/hip_runtime.h>

// ---------------------------------------------------------------------------
// Fused causal self-attention for MI455X (gfx1250, wave32, WMMA 16x16x32 f16)
//   B=16, S=2048, C=256, DK=64, DO=256
//   (0) pack W^T f16   (1) QKV projection GEMM (WMMA, Q pre-scaled by 1/8)
//   (2) flash attention: K/V tiles staged via global_load_async_to_lds_b128
//       (double-buffered, ASYNCcnt), WMMA scores + online softmax + WMMA PV
// ---------------------------------------------------------------------------

typedef _Float16 v8h  __attribute__((ext_vector_type(8)));
typedef _Float16 v16h __attribute__((ext_vector_type(16)));
typedef float    v8f  __attribute__((ext_vector_type(8)));
typedef float    v4f  __attribute__((ext_vector_type(4)));

#define NB  16
#define NS  2048
#define NC  256
#define NDK 64
#define NDO 256

__device__ __forceinline__ v16h cat16(v8h a, v8h b) {
  return __builtin_shufflevector(a, b, 0,1,2,3,4,5,6,7,8,9,10,11,12,13,14,15);
}
__device__ __forceinline__ v8h ldh8(const _Float16* p) { return *(const v8h*)p; }

// Async global -> LDS copy, 16 bytes per lane (gfx1250, tracked by ASYNCcnt).
// VDST = LDS byte address (low 32 bits of the generic shared pointer),
// ADDR = 64-bit global address VGPR pair.
__device__ __forceinline__ void async_b128(const _Float16* lds, const _Float16* g) {
  unsigned off = (unsigned)(size_t)lds;
  asm volatile("global_load_async_to_lds_b128 %0, %1, off"
               :: "v"(off), "v"(g) : "memory");
}

// Butterfly reductions across the 16-lane half-groups (wave32; xor masks <16
// keep lanes inside their half, matching the 16x16 C/D tile layout).
__device__ __forceinline__ float rowmax16(float v) {
  v = fmaxf(v, __shfl_xor(v, 1, 32));
  v = fmaxf(v, __shfl_xor(v, 2, 32));
  v = fmaxf(v, __shfl_xor(v, 4, 32));
  v = fmaxf(v, __shfl_xor(v, 8, 32));
  return v;
}
__device__ __forceinline__ float rowsum16(float v) {
  v += __shfl_xor(v, 1, 32);
  v += __shfl_xor(v, 2, 32);
  v += __shfl_xor(v, 4, 32);
  v += __shfl_xor(v, 8, 32);
  return v;
}

// ---------------------------------------------------------------------------
// Kernel 0: pack [Wq | Wk | Wv] transposed to f16:  wt[n][c], n in [0,384)
// ---------------------------------------------------------------------------
__global__ void wt_pack_kernel(const float* __restrict__ Wq,
                               const float* __restrict__ Wk,
                               const float* __restrict__ Wv,
                               _Float16* __restrict__ wt) {
  int i = blockIdx.x * 256 + threadIdx.x;   // [0, 384*256)
  int n = i >> 8;
  int c = i & 255;
  float v;
  if (n < 64)       v = Wq[c * 64 + n];
  else if (n < 128) v = Wk[c * 64 + (n - 64)];
  else              v = Wv[c * 256 + (n - 128)];
  wt[n * NC + c] = (_Float16)v;
}

// ---------------------------------------------------------------------------
// Kernel 1: QKV projection. GEMM [B*S x C] * [C x 384] via WMMA f16.
// ---------------------------------------------------------------------------
__global__ void __launch_bounds__(256) qkv_kernel(
    const float* __restrict__ x, const _Float16* __restrict__ wt,
    const float* __restrict__ bq, const float* __restrict__ bk,
    const float* __restrict__ bv,
    _Float16* __restrict__ qbuf, _Float16* __restrict__ kbuf,
    _Float16* __restrict__ vt) {
  const int w = threadIdx.x >> 5;
  const int lane = threadIdx.x & 31;
  const int ln = lane & 15;
  const int hi = lane >> 4;
  const int tile = blockIdx.x * 8 + w;          // 49152 tiles total
  const int ntile = tile % 24;
  const int mtile = tile / 24;
  const int row_base = mtile * 16;
  const int col_base = ntile * 16;

  v8f acc = {};
  const float*    xrow = x  + (size_t)(row_base + ln) * NC;
  const _Float16* wrow = wt + (size_t)(col_base + ln) * NC;

#pragma unroll
  for (int k0 = 0; k0 < NC; k0 += 32) {
    v4f xa0 = *(const v4f*)(xrow + k0 + hi * 8);
    v4f xa1 = *(const v4f*)(xrow + k0 + hi * 8 + 4);
    v4f xb0 = *(const v4f*)(xrow + k0 + 16 + hi * 8);
    v4f xb1 = *(const v4f*)(xrow + k0 + 16 + hi * 8 + 4);
    v16h a;
#pragma unroll
    for (int t = 0; t < 4; t++) {
      a[t]      = (_Float16)xa0[t];
      a[4 + t]  = (_Float16)xa1[t];
      a[8 + t]  = (_Float16)xb0[t];
      a[12 + t] = (_Float16)xb1[t];
    }
    v16h bf = cat16(ldh8(wrow + k0 + hi * 8), ldh8(wrow + k0 + 16 + hi * 8));
    acc = __builtin_amdgcn_wmma_f32_16x16x32_f16(false, a, false, bf,
                                                 (short)0, acc, false, false);
  }

  const int col = col_base + ln;
  if (col < 64) {
    float bias = bq[col];
#pragma unroll
    for (int r = 0; r < 8; r++) {
      int row = row_base + r + 8 * hi;
      qbuf[(size_t)row * NDK + col] = (_Float16)((acc[r] + bias) * 0.125f);
    }
  } else if (col < 128) {
    float bias = bk[col - 64];
#pragma unroll
    for (int r = 0; r < 8; r++) {
      int row = row_base + r + 8 * hi;
      kbuf[(size_t)row * NDK + (col - 64)] = (_Float16)(acc[r] + bias);
    }
  } else {
    float bias = bv[col - 128];
#pragma unroll
    for (int r = 0; r < 8; r++) {
      int row = row_base + r + 8 * hi;
      int bb = row >> 11, ss = row & 2047;
      vt[((size_t)bb * NDO + (col - 128)) * NS + ss] = (_Float16)(acc[r] + bias);
    }
  }
}

// ---------------------------------------------------------------------------
// Kernel 2: causal flash attention with async-LDS staged K/V tiles.
//   grid = (S/64, B); block = 256 threads = 8 waves.
//   wave w: q rows [blk*64 + (w&3)*16, +16), out cols [(w>>2)*128, +128)
//   Uniform per-block key loop (double-buffered async staging); waves past
//   their causal range skip compute but keep staging/barrier duties.
// ---------------------------------------------------------------------------
__global__ void __launch_bounds__(256, 1) attn_kernel(
    const _Float16* __restrict__ qbuf, const _Float16* __restrict__ kbuf,
    const _Float16* __restrict__ vt, float* __restrict__ out) {
  __shared__ __align__(16) _Float16 Kt[2][32 * 64];    // 2 x 4 KB  (keys x dk)
  __shared__ __align__(16) _Float16 Vt[2][256 * 32];   // 2 x 16 KB (do x keys)
  __shared__ __align__(16) _Float16 plds[8][16][32];   // 8 KB P staging

  const int tid = threadIdx.x;
  const int w = tid >> 5;
  const int lane = tid & 31;
  const int ln = lane & 15;
  const int hi = lane >> 4;
  const int qsub = w & 3;
  const int dhalf = w >> 2;
  const int b = blockIdx.y;
  const int qb_blk = blockIdx.x * 64;
  const int q_base = qb_blk + qsub * 16;
  const int kend = q_base + 15;      // this wave's last needed key
  const int jmax = qb_blk + 63;      // block's last needed key

  // Stage K tile (32x64 f16, contiguous 4KB: 1 async/thread) and V tile
  // (256 do-rows x 32 keys: 4 async/thread) for keys [j, j+32) into buffer.
  auto issue_tile = [&](int j, int buf) {
    const _Float16* kg = kbuf + (size_t)(b * NS + j) * NDK + tid * 8;
    async_b128(&Kt[buf][tid * 8], kg);
    const _Float16* vg = vt + (size_t)(b * NDO + tid) * NS + j;
#pragma unroll
    for (int c = 0; c < 4; c++)
      async_b128(&Vt[buf][tid * 32 + c * 8], vg + c * 8);
  };

  // Q fragments (d = 0..31 and 32..63), already scaled by 1/sqrt(dk)
  const _Float16* qrow = qbuf + (size_t)(b * NS + q_base + ln) * NDK;
  v16h aQ0 = cat16(ldh8(qrow + hi * 8),      ldh8(qrow + 16 + hi * 8));
  v16h aQ1 = cat16(ldh8(qrow + 32 + hi * 8), ldh8(qrow + 48 + hi * 8));

  float m[8], l[8];
  v8f acc[8];
#pragma unroll
  for (int r = 0; r < 8; r++) { m[r] = -3.0e38f; l[r] = 0.0f; acc[r] = (v8f){}; }

  issue_tile(0, 0);
  for (int j = 0; j <= jmax; j += 32) {
    const int buf = (j >> 5) & 1;
    // Current buffer's async copies done (own wave), then block-wide ready;
    // the same barrier proves everyone finished reading buffer buf^1.
    asm volatile("s_wait_asynccnt 0" ::: "memory");
    __syncthreads();
    if (j + 32 <= jmax) issue_tile(j + 32, buf ^ 1);

    if (j <= kend) {
      // ---- scores S = Q K^T for keys [j, j+32), B-frags from LDS ---------
      const _Float16* kr0 = &Kt[buf][(ln)      * NDK + hi * 8];
      const _Float16* kr1 = &Kt[buf][(16 + ln) * NDK + hi * 8];
      v16h b00 = cat16(ldh8(kr0),      ldh8(kr0 + 16));
      v16h b01 = cat16(ldh8(kr0 + 32), ldh8(kr0 + 48));
      v16h b10 = cat16(ldh8(kr1),      ldh8(kr1 + 16));
      v16h b11 = cat16(ldh8(kr1 + 32), ldh8(kr1 + 48));

      v8f s0 = {}, s1 = {};
      s0 = __builtin_amdgcn_wmma_f32_16x16x32_f16(false, aQ0, false, b00, (short)0, s0, false, false);
      s0 = __builtin_amdgcn_wmma_f32_16x16x32_f16(false, aQ1, false, b01, (short)0, s0, false, false);
      s1 = __builtin_amdgcn_wmma_f32_16x16x32_f16(false, aQ0, false, b10, (short)0, s1, false, false);
      s1 = __builtin_amdgcn_wmma_f32_16x16x32_f16(false, aQ1, false, b11, (short)0, s1, false, false);

      // ---- causal mask (only near the diagonal) -------------------------
      if (j + 31 > q_base) {
#pragma unroll
        for (int r = 0; r < 8; r++) {
          int row = q_base + r + 8 * hi;
          if (j + ln > row)      s0[r] = -1.0e30f;
          if (j + 16 + ln > row) s1[r] = -1.0e30f;
        }
      }

      // ---- online softmax + stage P into per-wave LDS -------------------
#pragma unroll
      for (int r = 0; r < 8; r++) {
        float tm = rowmax16(fmaxf(s0[r], s1[r]));
        float nm = fmaxf(m[r], tm);
        float sc = __expf(m[r] - nm);
        float p0 = __expf(s0[r] - nm);
        float p1 = __expf(s1[r] - nm);
        float rs = rowsum16(p0 + p1);
        l[r] = l[r] * sc + rs;
        m[r] = nm;
#pragma unroll
        for (int dt = 0; dt < 8; dt++) acc[dt][r] *= sc;
        plds[w][r + 8 * hi][ln]      = (_Float16)p0;
        plds[w][r + 8 * hi][16 + ln] = (_Float16)p1;
      }
      asm volatile("s_wait_dscnt 0" ::: "memory");   // cross-lane via LDS
      v16h aP = cat16(ldh8(&plds[w][ln][hi * 8]), ldh8(&plds[w][ln][16 + hi * 8]));

      // ---- O += P @ V, B-frags from LDS V tile --------------------------
#pragma unroll
      for (int dt = 0; dt < 8; dt++) {
        const _Float16* vr = &Vt[buf][(dhalf * 128 + dt * 16 + ln) * 32 + hi * 8];
        v16h bv16 = cat16(ldh8(vr), ldh8(vr + 16));
        acc[dt] = __builtin_amdgcn_wmma_f32_16x16x32_f16(false, aP, false, bv16,
                                                         (short)0, acc[dt], false, false);
      }
    }
  }

  // ---- epilogue: out = acc / l ------------------------------------------
#pragma unroll
  for (int r = 0; r < 8; r++) {
    float inv = 1.0f / l[r];
    int row = q_base + r + 8 * hi;
#pragma unroll
    for (int dt = 0; dt < 8; dt++) {
      int col = dhalf * 128 + dt * 16 + ln;
      out[(size_t)(b * NS + row) * NDO + col] = acc[dt][r] * inv;
    }
  }
}

// ---------------------------------------------------------------------------
extern "C" void kernel_launch(void* const* d_in, const int* in_sizes, int n_in,
                              void* d_out, int out_size, void* d_ws, size_t ws_size,
                              hipStream_t stream) {
  const float* x  = (const float*)d_in[0];
  const float* Wq = (const float*)d_in[1];
  const float* bq = (const float*)d_in[2];
  const float* Wk = (const float*)d_in[3];
  const float* bk = (const float*)d_in[4];
  const float* Wv = (const float*)d_in[5];
  const float* bv = (const float*)d_in[6];
  float* out = (float*)d_out;

  char* ws = (char*)d_ws;
  // ws layout (bytes): wt 196608 | qbuf 4 MB | kbuf 4 MB | vt 16 MB  (~24.2 MB)
  _Float16* wt   = (_Float16*)(ws);
  _Float16* qbuf = (_Float16*)(ws + 196608);
  _Float16* kbuf = (_Float16*)(ws + 196608 + 4194304);
  _Float16* vt   = (_Float16*)(ws + 196608 + 8388608);

  wt_pack_kernel<<<384, 256, 0, stream>>>(Wq, Wk, Wv, wt);
  qkv_kernel<<<6144, 256, 0, stream>>>(x, wt, bq, bk, bv, qbuf, kbuf, vt);
  attn_kernel<<<dim3(NS / 64, NB), 256, 0, stream>>>(qbuf, kbuf, vt, out);
}
